// question_layer_LSTM_72241349919244
// MI455X (gfx1250) — matrix operand order repaired and checked by reading, not compiled, not run on hardware
//
#include <hip/hip_runtime.h>

// ---------------------------------------------------------------------------
// Types for CDNA5 WMMA (gfx1250, wave32)
// ---------------------------------------------------------------------------
typedef __attribute__((ext_vector_type(16))) __bf16 v16bf;
typedef __attribute__((ext_vector_type(8)))  float  v8f;

#define LDS_STRIDE 48   // 32 bf16 payload + 16 pad; 96B rows keep 16B alignment

__device__ __forceinline__ unsigned short f2bf(float f) {
    unsigned int u = __float_as_uint(f);
    unsigned int r = (u + 0x7FFFu + ((u >> 16) & 1u)) >> 16;
    return (unsigned short)r;
}

// ---------------------------------------------------------------------------
// Weight transpose + fp32 -> bf16 convert:  wt[n*Kpad + k] = bf16(w[k*N + n])
// ---------------------------------------------------------------------------
__global__ void k_wtrans_bf16(const float* __restrict__ w,
                              unsigned short* __restrict__ wt,
                              int K, int N, int Kpad) {
    long long idx = (long long)blockIdx.x * blockDim.x + threadIdx.x;
    long long total = (long long)N * Kpad;
    if (idx >= total) return;
    int k = (int)(idx % Kpad);
    int n = (int)(idx / Kpad);
    float v = (k < K) ? w[(size_t)k * N + n] : 0.f;
    wt[idx] = f2bf(v);
}

// ---------------------------------------------------------------------------
// Embedding gather -> bf16 A matrix, row (t*B + b), K padded 300 -> 320
// ---------------------------------------------------------------------------
__global__ void k_embed_bf16(const int* __restrict__ tok,
                             const float* __restrict__ emb,
                             unsigned short* __restrict__ xbf,
                             int B, int T, int E, int Kpad) {
    long long idx = (long long)blockIdx.x * blockDim.x + threadIdx.x;
    long long total = (long long)T * B * Kpad;
    if (idx >= total) return;
    int e = (int)(idx % Kpad);
    long long rb = idx / Kpad;
    int b = (int)(rb % B);
    int t = (int)(rb / B);
    float v = 0.f;
    if (e < E) {
        int token = tok[(size_t)b * T + t];
        v = emb[(size_t)token * E + e];
    }
    xbf[idx] = f2bf(v);
}

// ---------------------------------------------------------------------------
// bf16 WMMA GEMM:  C[M][N] = A[M][K] * Bt[N][K]^T  (+ bias, optional tanh)
// block = 256 threads (8 waves); (MT*64) x 128 tile; BK = 32, double-buffered
// waves arranged 4 (m) x 2 (n); each wave owns MT x 4 WMMA 16x16 tiles
//   MT=2 -> 128x128 tile (big batched projections)
//   MT=1 -> 64x128 tile  (recurrent step: 2x more workgroups on critical path)
// ---------------------------------------------------------------------------
template <int MT, int BIAS, int ACT>
__global__ __launch_bounds__(256) void k_gemm_bf16(
        const unsigned short* __restrict__ A,
        const unsigned short* __restrict__ Bt,
        const float* __restrict__ bias,
        float* __restrict__ C,
        int M, int N, int K) {
    constexpr int BM = MT * 64;
    __shared__ unsigned short sA[2][BM * LDS_STRIDE];
    __shared__ unsigned short sB[2][128 * LDS_STRIDE];

    const int tid   = threadIdx.x;
    const int wave  = tid >> 5;
    const int lane  = tid & 31;
    const int lrow  = lane & 15;
    const int khalf = lane >> 4;

    const int block_m = blockIdx.y * BM;
    const int block_n = blockIdx.x * 128;

    const int wm = (wave >> 1) * (MT * 16);  // 4 wave-rows
    const int wn = (wave & 1) * 64;          // 2 wave-cols

    v8f acc[MT][4];
    const v8f vzero = {0.f, 0.f, 0.f, 0.f, 0.f, 0.f, 0.f, 0.f};
#pragma unroll
    for (int i = 0; i < MT; ++i)
#pragma unroll
        for (int j = 0; j < 4; ++j) acc[i][j] = vzero;

    union FragU { v16bf v; uint4 u4[2]; };

    // Stage one 32-wide K-slab of A (BM x 32) and Bt (128 x 32) into LDS buf.
    auto stage = [&](int buf, int k0) {
        constexpr int ACH = (BM * 4) / 256;  // 16B chunks of A per thread
#pragma unroll
        for (int i = 0; i < ACH; ++i) {
            int ch = tid + i * 256;
            int r = ch >> 2, c4 = ch & 3;
            *(uint4*)(&sA[buf][r * LDS_STRIDE + c4 * 8]) =
                *(const uint4*)(A + (size_t)(block_m + r) * K + k0 + c4 * 8);
        }
#pragma unroll
        for (int i = 0; i < 2; ++i) {        // 512 B-chunks / 256 threads
            int ch = tid + i * 256;
            int r = ch >> 2, c4 = ch & 3;
            *(uint4*)(&sB[buf][r * LDS_STRIDE + c4 * 8]) =
                *(const uint4*)(Bt + (size_t)(block_n + r) * K + k0 + c4 * 8);
        }
    };

    const int niter = K >> 5;  // K / 32 (all K here are multiples of 32)
    stage(0, 0);
    __syncthreads();

    for (int it = 0; it < niter; ++it) {
        const int cur = it & 1;
        if (it + 1 < niter) stage(cur ^ 1, (it + 1) * 32);

        FragU af[MT], bfr[4];
        // A fragment (16-bit 16x32 layout): lanes 0-15 M=lrow;
        // elems 0..7 -> K = khalf*8 + 0..7 ; elems 8..15 -> K = 16+khalf*8+0..7
#pragma unroll
        for (int mi = 0; mi < MT; ++mi) {
            const unsigned short* base = &sA[cur][(wm + mi * 16 + lrow) * LDS_STRIDE];
            af[mi].u4[0] = *(const uint4*)(base + khalf * 8);
            af[mi].u4[1] = *(const uint4*)(base + 16 + khalf * 8);
        }
        // B fragment: N = lrow of tile, same K order as A (khalf*8 + 0..7, 16+khalf*8 + 0..7)
#pragma unroll
        for (int ni = 0; ni < 4; ++ni) {
            const unsigned short* base = &sB[cur][(wn + ni * 16 + lrow) * LDS_STRIDE];
            bfr[ni].u4[0] = *(const uint4*)(base + khalf * 8);
            bfr[ni].u4[1] = *(const uint4*)(base + 16 + khalf * 8);
        }
#pragma unroll
        for (int mi = 0; mi < MT; ++mi)
#pragma unroll
            for (int ni = 0; ni < 4; ++ni)
                acc[mi][ni] = __builtin_amdgcn_wmma_f32_16x16x32_bf16(
                    false, af[mi].v, false, bfr[ni].v, (short)0, acc[mi][ni],
                    false, false);

        __syncthreads();  // reads of 'cur' and writes of 'cur^1' both done
    }

    // Epilogue: C/D layout — lane group khalf: rows v + 8*khalf, col = lrow
#pragma unroll
    for (int mi = 0; mi < MT; ++mi) {
#pragma unroll
        for (int ni = 0; ni < 4; ++ni) {
            int col = block_n + wn + ni * 16 + lrow;
            float bv = BIAS ? bias[col] : 0.f;
#pragma unroll
            for (int v = 0; v < 8; ++v) {
                int row = block_m + wm + mi * 16 + v + khalf * 8;
                float x = acc[mi][ni][v] + bv;
                if (ACT) x = tanhf(x);
                C[(size_t)row * N + col] = x;
            }
        }
    }
}

// ---------------------------------------------------------------------------
// LSTM cell (keras gate order i,f,g,o):  z = xp[t] + h@U (hu precomputed)
// ---------------------------------------------------------------------------
__global__ void k_lstm_cell(const float* __restrict__ xp,   // [T*B][4H], row t*B+b
                            const float* __restrict__ hu,   // [B][4H]
                            float* __restrict__ cst,        // [B][H]
                            float* __restrict__ hs_out,     // [T*B][H] or null
                            float* __restrict__ h_out,      // [B][H]   or null
                            unsigned short* __restrict__ hbf, // [B][H] bf16
                            int t, int B, int H) {
    int idx = blockIdx.x * blockDim.x + threadIdx.x;
    if (idx >= B * H) return;
    int b = idx / H, k = idx % H;
    const float* zrow = xp + (size_t)(t * B + b) * (4 * H);
    const float* hrow = hu + (size_t)b * (4 * H);
    float zi = zrow[k]         + hrow[k];
    float zf = zrow[H + k]     + hrow[H + k];
    float zg = zrow[2 * H + k] + hrow[2 * H + k];
    float zo = zrow[3 * H + k] + hrow[3 * H + k];
    float ig = 1.f / (1.f + __expf(-zi));
    float fg = 1.f / (1.f + __expf(-zf));
    float gg = tanhf(zg);
    float og = 1.f / (1.f + __expf(-zo));
    float c  = fg * cst[idx] + ig * gg;
    cst[idx] = c;
    float h  = og * tanhf(c);
    if (hs_out) hs_out[(size_t)(t * B + b) * H + k] = h;
    if (h_out)  h_out[idx] = h;
    hbf[idx] = f2bf(h);
}

// ---------------------------------------------------------------------------
// BatchNorm (center=False, scale=False) fused with bf16 convert
// ---------------------------------------------------------------------------
__global__ void k_bn_bf16(const float* __restrict__ x,
                          const float* __restrict__ mean,
                          const float* __restrict__ var,
                          unsigned short* __restrict__ ybf,
                          long long total, int H) {
    long long idx = (long long)blockIdx.x * blockDim.x + threadIdx.x;
    if (idx >= total) return;
    int k = (int)(idx % H);
    float y = (x[idx] - mean[k]) * rsqrtf(var[k] + 1e-3f);
    ybf[idx] = f2bf(y);
}

__global__ void k_zero_state(unsigned short* __restrict__ hbf,
                             float* __restrict__ c, int n) {
    int idx = blockIdx.x * blockDim.x + threadIdx.x;
    if (idx < n) { hbf[idx] = 0; c[idx] = 0.f; }
}

// ---------------------------------------------------------------------------
// Launch
// ---------------------------------------------------------------------------
extern "C" void kernel_launch(void* const* d_in, const int* in_sizes, int n_in,
                              void* d_out, int out_size, void* d_ws, size_t ws_size,
                              hipStream_t stream) {
    (void)in_sizes; (void)n_in; (void)out_size; (void)ws_size;
    const int*   tokens = (const int*)d_in[0];
    const float* emb    = (const float*)d_in[1];
    const float* w1     = (const float*)d_in[2];
    const float* u1     = (const float*)d_in[3];
    const float* b1     = (const float*)d_in[4];
    const float* mean1  = (const float*)d_in[5];
    const float* var1   = (const float*)d_in[6];
    const float* w2     = (const float*)d_in[7];
    const float* u2     = (const float*)d_in[8];
    const float* b2     = (const float*)d_in[9];
    const float* mean2  = (const float*)d_in[10];
    const float* var2   = (const float*)d_in[11];
    const float* wd     = (const float*)d_in[12];
    const float* bd     = (const float*)d_in[13];
    float* out = (float*)d_out;

    const int B = 256, T = 26, E = 300, Epad = 320, H = 1024, G = 4096;
    const long long M1 = (long long)T * B;  // 6656

    char* ws = (char*)d_ws;
    size_t off = 0;
    auto alloc = [&](size_t bytes) {
        size_t p = off; off = (off + bytes + 255) & ~(size_t)255; return p;
    };
    unsigned short* w1t  = (unsigned short*)(ws + alloc((size_t)G * Epad * 2));
    unsigned short* u1t  = (unsigned short*)(ws + alloc((size_t)G * H * 2));
    unsigned short* w2t  = (unsigned short*)(ws + alloc((size_t)G * H * 2));
    unsigned short* u2t  = (unsigned short*)(ws + alloc((size_t)G * H * 2));
    unsigned short* wdt  = (unsigned short*)(ws + alloc((size_t)H * H * 2));
    unsigned short* xbf  = (unsigned short*)(ws + alloc((size_t)M1 * Epad * 2));
    float*          xp   = (float*)(ws + alloc((size_t)M1 * G * 4));  // reused L1/L2
    float*          hs   = (float*)(ws + alloc((size_t)M1 * H * 4));
    unsigned short* hsbn = (unsigned short*)(ws + alloc((size_t)M1 * H * 2));
    float*          hu   = (float*)(ws + alloc((size_t)B * G * 4));
    float*          cst  = (float*)(ws + alloc((size_t)B * H * 4));
    unsigned short* hbf  = (unsigned short*)(ws + alloc((size_t)B * H * 2));
    float*          h2   = (float*)(ws + alloc((size_t)B * H * 4));
    unsigned short* hTbn = (unsigned short*)(ws + alloc((size_t)B * H * 2));

    dim3 blk(256);
    auto blocks = [](long long n) { return dim3((unsigned)((n + 255) / 256)); };

    // Weight prep (transpose + bf16)
    k_wtrans_bf16<<<blocks((long long)G * Epad), blk, 0, stream>>>(w1, w1t, E, G, Epad);
    k_wtrans_bf16<<<blocks((long long)G * H), blk, 0, stream>>>(u1, u1t, H, G, H);
    k_wtrans_bf16<<<blocks((long long)G * H), blk, 0, stream>>>(w2, w2t, H, G, H);
    k_wtrans_bf16<<<blocks((long long)G * H), blk, 0, stream>>>(u2, u2t, H, G, H);
    k_wtrans_bf16<<<blocks((long long)H * H), blk, 0, stream>>>(wd, wdt, H, H, H);

    // Embedding gather -> bf16
    k_embed_bf16<<<blocks(M1 * Epad), blk, 0, stream>>>(tokens, emb, xbf, B, T, E, Epad);

    // xp1 = x @ w1 + b1   [6656 x 4096], 128x128 tiles (52x32 blocks)
    k_gemm_bf16<2, 1, 0><<<dim3(G / 128, (unsigned)(M1 / 128)), blk, 0, stream>>>(
        xbf, w1t, b1, xp, (int)M1, G, Epad);

    // LSTM layer 1 (return sequences); recurrent GEMM uses 64x128 tiles
    // (4x32 = 128 workgroups on the serial critical path)
    k_zero_state<<<blocks(B * H), blk, 0, stream>>>(hbf, cst, B * H);
    for (int t = 0; t < T; ++t) {
        k_gemm_bf16<1, 0, 0><<<dim3(G / 128, B / 64), blk, 0, stream>>>(
            hbf, u1t, nullptr, hu, B, G, H);
        k_lstm_cell<<<blocks(B * H), blk, 0, stream>>>(
            xp, hu, cst, hs, nullptr, hbf, t, B, H);
    }

    // BN1 -> bf16
    k_bn_bf16<<<blocks(M1 * H), blk, 0, stream>>>(hs, mean1, var1, hsbn, M1 * H, H);

    // xp2 = bn(hs) @ w2 + b2   [6656 x 4096]  (reuses xp buffer)
    k_gemm_bf16<2, 1, 0><<<dim3(G / 128, (unsigned)(M1 / 128)), blk, 0, stream>>>(
        hsbn, w2t, b2, xp, (int)M1, G, H);

    // LSTM layer 2 (return last state)
    k_zero_state<<<blocks(B * H), blk, 0, stream>>>(hbf, cst, B * H);
    for (int t = 0; t < T; ++t) {
        k_gemm_bf16<1, 0, 0><<<dim3(G / 128, B / 64), blk, 0, stream>>>(
            hbf, u2t, nullptr, hu, B, G, H);
        k_lstm_cell<<<blocks(B * H), blk, 0, stream>>>(
            xp, hu, cst, nullptr, h2, hbf, t, B, H);
    }

    // BN2 -> bf16
    k_bn_bf16<<<blocks((long long)B * H), blk, 0, stream>>>(
        h2, mean2, var2, hTbn, (long long)B * H, H);

    // out = tanh(bn(hT) @ wd + bd)   [256 x 1024], 64x128 tiles (8x4 blocks)
    k_gemm_bf16<1, 1, 1><<<dim3(H / 128, B / 64), blk, 0, stream>>>(
        hTbn, wdt, bd, out, B, H, H);
}
